// SpatialAttention2d_59390807769337
// MI455X (gfx1250) — matrix operand order repaired and checked by reading, not compiled, not run on hardware
//
#include <hip/hip_runtime.h>
#include <hip/hip_bf16.h>

// ---------------------------------------------------------------------------
// SpatialAttention2d fused kernels for gfx1250 (MI455X), wave32 + WMMA bf16.
//   B=8, C=512, H=W=64 -> N=4096, CQK=64
// Stage 1: qkv projection GEMMs (fp32 in -> bf16 Q/K/V in workspace)
// Stage 2: flash-attention (online softmax), bf16 WMMA, fp32 accum,
//          double-buffered K/V tiles via GLOBAL_LOAD_ASYNC_TO_LDS (ASYNCcnt),
//          age-split s_wait_asynccnt so the prefetch pipe never drains,
//          epilogue: out = alpha * (P V^T)/l + x
// ---------------------------------------------------------------------------

typedef __attribute__((ext_vector_type(16))) __bf16 v16bf;
typedef __attribute__((ext_vector_type(8)))  float  v8f;
typedef int vec4i __attribute__((vector_size(16)));  // matches builtin param type

#define DEV __device__ __forceinline__

constexpr int kBATCH = 8;
constexpr int kC     = 512;
constexpr int kN     = 4096;
constexpr int kCQK   = 64;

#if defined(__has_builtin)
#  if __has_builtin(__builtin_amdgcn_global_load_async_to_lds_b128)
#    define HAVE_ASYNC_LDS 1
#  endif
#  if __has_builtin(__builtin_amdgcn_s_wait_asynccnt)
#    define HAVE_ASYNC_WAIT 1
#  endif
#endif

DEV v16bf pack16(uint4 lo, uint4 hi) {
  union { uint4 q[2]; v16bf v; } u;
  u.q[0] = lo; u.q[1] = hi;
  return u.v;
}

DEV v8f zero8() { v8f z = {0.f,0.f,0.f,0.f,0.f,0.f,0.f,0.f}; return z; }

// A-fragment (16x32 bf16), source laid out row-major [m][k], stride in elems.
// ISA 7.12.2: lanes 0-15 row M=lane, K={0..7,16..23}; lanes 16-31 K={8..15,24..31}.
DEV v16bf load_a16(const __bf16* base, int m0, int k0, int stride, int lane) {
  const int half = lane >> 4, mr = lane & 15;
  const __bf16* p = base + (size_t)(m0 + mr) * stride;
  const int kA = k0 + half * 8;
  const int kB2 = k0 + 16 + half * 8;
  return pack16(*(const uint4*)(p + kA), *(const uint4*)(p + kB2));
}

// B-fragment (32x16 bf16), source laid out col-major-by-n [n][k], stride elems.
// lanes 0-15: column N=lane, K={0..7,16..23}; lanes 16-31: K={8..15,24..31}.
DEV v16bf load_b16(const __bf16* base, int n0, int k0, int stride, int lane) {
  const int half = lane >> 4, nc = lane & 15;
  const __bf16* p = base + (size_t)(n0 + nc) * stride + k0 + half * 8;
  return pack16(*(const uint4*)p, *(const uint4*)(p + 16));
}

// A-fragment from fp32 row-major weights, converted to bf16.
DEV v16bf load_a_f32(const float* w, int row0, int k0, int stride, int lane) {
  const int half = lane >> 4, mr = lane & 15;
  const float* p = w + (size_t)(row0 + mr) * stride;
  const int kA = k0 + half * 8;
  const int kB2 = k0 + 16 + half * 8;
  float4 f0 = *(const float4*)(p + kA);
  float4 f1 = *(const float4*)(p + kA + 4);
  float4 f2 = *(const float4*)(p + kB2);
  float4 f3 = *(const float4*)(p + kB2 + 4);
  v16bf a;
  a[0]=(__bf16)f0.x;  a[1]=(__bf16)f0.y;  a[2]=(__bf16)f0.z;  a[3]=(__bf16)f0.w;
  a[4]=(__bf16)f1.x;  a[5]=(__bf16)f1.y;  a[6]=(__bf16)f1.z;  a[7]=(__bf16)f1.w;
  a[8]=(__bf16)f2.x;  a[9]=(__bf16)f2.y;  a[10]=(__bf16)f2.z; a[11]=(__bf16)f2.w;
  a[12]=(__bf16)f3.x; a[13]=(__bf16)f3.y; a[14]=(__bf16)f3.z; a[15]=(__bf16)f3.w;
  return a;
}

DEV v8f wmma_bf16(v16bf a, v16bf b, v8f c) {
  return __builtin_amdgcn_wmma_f32_16x16x32_bf16(
      /*neg_a=*/false, a, /*neg_b=*/false, b,
      /*c_mod=*/(short)0, c, /*reuse_a=*/false, /*reuse_b=*/false);
}

// Reductions across each 16-lane half of the wave32 (xor masks < 16 stay in-half).
DEV float redmax16(float v) {
  #pragma unroll
  for (int m = 1; m < 16; m <<= 1) v = fmaxf(v, __shfl_xor(v, m, 32));
  return v;
}
DEV float redsum16(float v) {
  #pragma unroll
  for (int m = 1; m < 16; m <<= 1) v += __shfl_xor(v, m, 32);
  return v;
}

// One 16-byte global->LDS transfer; async (ASYNCcnt) when the toolchain has it.
DEV void gl_async_b128(const __bf16* g, __bf16* l) {
#if defined(HAVE_ASYNC_LDS)
  __builtin_amdgcn_global_load_async_to_lds_b128(
      (__attribute__((address_space(1))) vec4i*)g,
      (__attribute__((address_space(3))) vec4i*)l, 0, 0);
#else
  *(uint4*)l = *(const uint4*)g;
#endif
}

// Age-based waits; async loads complete in order (ISA 10.x async ordering),
// so waiting ASYNCcnt <= n means "all but the n newest transfers have landed".
template <short N>
DEV void async_wait_le() {
#if defined(HAVE_ASYNC_WAIT)
  __builtin_amdgcn_s_wait_asynccnt(N);
#endif
}

// ---------------------------------------------------------------------------
// Stage 1: Q/K/V projection.  grid=(N/32, B), block=256 (8 waves).
// Each wave owns 5 of the 40 output M-blocks x 2 N-blocks (acc = 80 VGPRs,
// keeps the kernel under 256 VGPRs -> no s_set_vgpr_msb churn in the loop).
// Outputs: Qb,Kb = [b][n][64] bf16 ; Vb = [b][c][n] bf16.
// ---------------------------------------------------------------------------
__global__ __launch_bounds__(256) void qkv_kernel(
    const float* __restrict__ x,
    const float* __restrict__ wq, const float* __restrict__ bq,
    const float* __restrict__ wk, const float* __restrict__ bk,
    const float* __restrict__ wv, const float* __restrict__ bv,
    __bf16* __restrict__ Qb, __bf16* __restrict__ Kb, __bf16* __restrict__ Vb) {
  extern __shared__ char smem[];
  __bf16* xT = (__bf16*)smem;  // [32 n][512 ci] bf16 = 32KB

  const int b = blockIdx.y;
  const int n0 = blockIdx.x * 32;
  const int t = threadIdx.x;
  const int wave = t >> 5, lane = t & 31;
  const int half = lane >> 4, col = lane & 15;
  const float* xb = x + (size_t)b * kC * kN + n0;

  // Transpose-load x tile into LDS as bf16 [n][ci].
  #pragma unroll
  for (int it = 0; it < 16; ++it) {
    int slot = it * 256 + t;          // 512 ci-rows x 8 float4
    int ci = slot >> 3, q4 = (slot & 7) * 4;
    float4 f = *(const float4*)(xb + (size_t)ci * kN + q4);
    xT[(q4 + 0) * kC + ci] = (__bf16)f.x;
    xT[(q4 + 1) * kC + ci] = (__bf16)f.y;
    xT[(q4 + 2) * kC + ci] = (__bf16)f.z;
    xT[(q4 + 3) * kC + ci] = (__bf16)f.w;
  }
  __syncthreads();

  // 40 output M-blocks total (q:0-3, k:4-7, v:8-39); wave owns 5.
  v8f acc[5][2];
  #pragma unroll
  for (int i = 0; i < 5; ++i)
    #pragma unroll
    for (int j = 0; j < 2; ++j) acc[i][j] = zero8();

  for (int kc = 0; kc < kC; kc += 32) {
    v16bf a[5];
    #pragma unroll
    for (int i = 0; i < 5; ++i) {
      int mb = wave * 5 + i;
      const float* w; int row;
      if (mb < 4)       { w = wq; row = mb * 16; }
      else if (mb < 8)  { w = wk; row = (mb - 4) * 16; }
      else              { w = wv; row = (mb - 8) * 16; }
      a[i] = load_a_f32(w, row, kc, kC, lane);
    }
    #pragma unroll
    for (int j = 0; j < 2; ++j) {
      v16bf bx = load_b16(xT, j * 16, kc, kC, lane);
      #pragma unroll
      for (int i = 0; i < 5; ++i) acc[i][j] = wmma_bf16(a[i], bx, acc[i][j]);
    }
  }

  // Epilogue: +bias, convert, scatter to workspace layouts.
  #pragma unroll
  for (int i = 0; i < 5; ++i) {
    int mb = wave * 5 + i;
    const float* bias; int row; int kind;
    if (mb < 4)       { bias = bq; row = mb * 16;       kind = 0; }
    else if (mb < 8)  { bias = bk; row = (mb - 4) * 16; kind = 1; }
    else              { bias = bv; row = (mb - 8) * 16; kind = 2; }
    float br[8];
    #pragma unroll
    for (int r = 0; r < 8; ++r) br[r] = bias[row + r + half * 8];
    #pragma unroll
    for (int j = 0; j < 2; ++j) {
      int n = n0 + j * 16 + col;
      #pragma unroll
      for (int r = 0; r < 8; ++r) {
        int c = row + r + half * 8;
        __bf16 v = (__bf16)(acc[i][j][r] + br[r]);
        if (kind == 0)      Qb[((size_t)b * kN + n) * kCQK + c] = v;
        else if (kind == 1) Kb[((size_t)b * kN + n) * kCQK + c] = v;
        else                Vb[((size_t)b * kC + c) * kN + n] = v;
      }
    }
  }
}

// ---------------------------------------------------------------------------
// Stage 2: flash attention. grid=(N/64, B), block=256 (8 waves).
// Double-buffered K/V tiles; async copies overlap both GEMM phases.
// LDS map (dynamic, 156160 B):
//   buf0: V [0,64K) K [64K,72K) | buf1: V [72K,136K) K [136K,144K)
//   Pl [144K,152K) | cfl @152K | ll @152K+256 | Ot (epilogue) reuses [0,128K)
// ---------------------------------------------------------------------------
constexpr int kBufStride = 73728;   // 72KB per buffer
constexpr int kKOff      = 65536;   // K tile offset within a buffer

// Issue one step's K (2 b128s/thread, FIRST) then V (16 b128s/thread).
// Issue order matters: waits below rely on in-order async completion.
DEV void issue_tiles(const __bf16* Ksrc, const __bf16* Vsrc,
                     __bf16* Kl, __bf16* Vl, int t) {
  #pragma unroll
  for (int it = 0; it < 2; ++it) {
    int slot = it * 256 + t, s = slot >> 3, o = (slot & 7) * 8;
    gl_async_b128(Ksrc + s * kCQK + o, Kl + s * 64 + o);
  }
  #pragma unroll
  for (int it = 0; it < 16; ++it) {
    int slot = it * 256 + t, c = slot >> 3, o = (slot & 7) * 8;
    gl_async_b128(Vsrc + (size_t)c * kN + o, Vl + c * 64 + o);
  }
}

__global__ __launch_bounds__(256) void attn_kernel(
    const float* __restrict__ x,
    const __bf16* __restrict__ Qb, const __bf16* __restrict__ Kb,
    const __bf16* __restrict__ Vb,
    const float* __restrict__ alpha_p, float* __restrict__ out) {
  extern __shared__ char smem[];
  __bf16* Pl  = (__bf16*)(smem + 147456);        // [64 q][64 s] = 8KB
  float*  cfl = (float*)(smem + 155648);         // [64] correction factors
  float*  ll  = (float*)(smem + 155648 + 256);   // [64] row sums
  float*  Ot  = (float*)smem;                    // epilogue reuse: [512][64] f32

  const int b = blockIdx.y;
  const int q0 = blockIdx.x * 64;
  const int t = threadIdx.x;
  const int wave = t >> 5, lane = t & 31;
  const int half = lane >> 4, col = lane & 15;
  const int c0 = wave * 64;  // this wave's channel chunk

  const __bf16* Qt = Qb + ((size_t)b * kN + q0) * kCQK;
  const __bf16* Kbase = Kb + (size_t)b * kN * kCQK;
  const __bf16* Vbase = Vb + (size_t)b * kC * kN;

  v16bf qa0, qa1;
  if (wave < 4) {  // Q A-fragments are loop-invariant: keep in registers
    qa0 = load_a16(Qt, wave * 16, 0,  kCQK, lane);
    qa1 = load_a16(Qt, wave * 16, 32, kCQK, lane);
  }

  float mrow[8], lrow[8];
  #pragma unroll
  for (int r = 0; r < 8; ++r) { mrow[r] = -3.0e38f; lrow[r] = 0.f; }

  v8f acc[4][4];  // [query M-block][channel N-block], 128 VGPRs of f32 accum
  #pragma unroll
  for (int i = 0; i < 4; ++i)
    #pragma unroll
    for (int j = 0; j < 4; ++j) acc[i][j] = zero8();

  // Prologue: prefetch step 0 into buffer 0.
  issue_tiles(Kbase, Vbase, (__bf16*)(smem + kKOff), (__bf16*)smem, t);

  int buf = 0;
  for (int s0 = 0; s0 < kN; s0 += 64, buf ^= 1) {
    // Outstanding (oldest first): K_cur(2), V_cur(16) [, next step's 18].
    // Wait <=16: K_cur landed; V_cur may still fly while we do scores.
    async_wait_le<16>();
    __syncthreads();

    // Kick off the next step's tiles into the other buffer (overlaps compute).
    bool has_next = (s0 + 64 < kN);
    if (has_next) {
      char* nb = smem + (buf ^ 1) * kBufStride;
      issue_tiles(Kbase + (size_t)(s0 + 64) * kCQK, Vbase + (s0 + 64),
                  (__bf16*)(nb + kKOff), (__bf16*)nb, t);
    }

    const __bf16* Vl = (const __bf16*)(smem + buf * kBufStride);
    const __bf16* Kl = (const __bf16*)(smem + buf * kBufStride + kKOff);

    // ---- phase 1: scores S = Q^T K, online softmax (waves 0-3) ----
    if (wave < 4) {
      v8f sc[4];
      #pragma unroll
      for (int j = 0; j < 4; ++j) sc[j] = zero8();
      #pragma unroll
      for (int j = 0; j < 4; ++j) {
        sc[j] = wmma_bf16(qa0, load_b16(Kl, j * 16, 0,  64, lane), sc[j]);
        sc[j] = wmma_bf16(qa1, load_b16(Kl, j * 16, 32, 64, lane), sc[j]);
      }
      #pragma unroll
      for (int r = 0; r < 8; ++r) {
        float mx = fmaxf(fmaxf(sc[0][r], sc[1][r]), fmaxf(sc[2][r], sc[3][r]));
        mx = redmax16(mx);
        float mnew = fmaxf(mrow[r], mx);
        float cf = __expf(mrow[r] - mnew);
        mrow[r] = mnew;
        int row = wave * 16 + r + half * 8;
        float rs = 0.f;
        #pragma unroll
        for (int j = 0; j < 4; ++j) {
          float p = __expf(sc[j][r] - mnew);
          rs += p;
          Pl[row * 64 + j * 16 + col] = (__bf16)p;  // D-layout -> A-layout via LDS
        }
        rs = redsum16(rs);
        lrow[r] = lrow[r] * cf + rs;
        if (col == 0) cfl[row] = cf;
      }
    }
    // V_cur must be fully landed before phase 2.  If we issued 18 newer
    // transfers, waiting <=18 drains exactly through V_cur; otherwise drain all.
    if (has_next) async_wait_le<18>(); else async_wait_le<0>();
    __syncthreads();

    // ---- phase 2: rescale + O += P * V^T (all 8 waves) ----
    float cfr[4][8];
    #pragma unroll
    for (int i = 0; i < 4; ++i)
      #pragma unroll
      for (int r = 0; r < 8; ++r) cfr[i][r] = cfl[i * 16 + r + half * 8];
    #pragma unroll
    for (int i = 0; i < 4; ++i)
      #pragma unroll
      for (int j = 0; j < 4; ++j)
        #pragma unroll
        for (int r = 0; r < 8; ++r) acc[i][j][r] *= cfr[i][r];

    #pragma unroll
    for (int kc = 0; kc < 64; kc += 32) {
      v16bf pa[4];
      #pragma unroll
      for (int i = 0; i < 4; ++i) pa[i] = load_a16(Pl, i * 16, kc, 64, lane);
      #pragma unroll
      for (int j = 0; j < 4; ++j) {
        v16bf bvf = load_b16(Vl, c0 + j * 16, kc, 64, lane);
        #pragma unroll
        for (int i = 0; i < 4; ++i) acc[i][j] = wmma_bf16(pa[i], bvf, acc[i][j]);
      }
    }
    // No trailing barrier: next iteration's top barrier separates Pl/Kl/Vl reuse.
  }
  __syncthreads();

  // publish row sums
  if (wave < 4 && col == 0) {
    #pragma unroll
    for (int r = 0; r < 8; ++r) ll[wave * 16 + r + half * 8] = lrow[r];
  }
  __syncthreads();

  const float alpha = alpha_p[0];
  float lr[4][8];
  #pragma unroll
  for (int i = 0; i < 4; ++i)
    #pragma unroll
    for (int r = 0; r < 8; ++r) lr[i][r] = ll[i * 16 + r + half * 8];

  // stage normalized O into LDS transposed [c][q] (reuses tile region)
  #pragma unroll
  for (int i = 0; i < 4; ++i)
    #pragma unroll
    for (int j = 0; j < 4; ++j) {
      int c = c0 + j * 16 + col;
      #pragma unroll
      for (int r = 0; r < 8; ++r) {
        int q = i * 16 + r + half * 8;
        Ot[c * 64 + q] = alpha * acc[i][j][r] / lr[i][r];
      }
    }
  __syncthreads();

  // coalesced write: out[b][c][q0..q0+63] = Ot + x
  const float* xb = x + (size_t)b * kC * kN + q0;
  float* ob = out + (size_t)b * kC * kN + q0;
  #pragma unroll
  for (int it = 0; it < 32; ++it) {
    int slot = it * 256 + t;
    int c = slot >> 4, qq = (slot & 15) * 4;
    float4 xv = *(const float4*)(xb + (size_t)c * kN + qq);
    float4 ov;
    ov.x = Ot[c * 64 + qq + 0] + xv.x;
    ov.y = Ot[c * 64 + qq + 1] + xv.y;
    ov.z = Ot[c * 64 + qq + 2] + xv.z;
    ov.w = Ot[c * 64 + qq + 3] + xv.w;
    *(float4*)(ob + (size_t)c * kN + qq) = ov;
  }
}

// ---------------------------------------------------------------------------
extern "C" void kernel_launch(void* const* d_in, const int* in_sizes, int n_in,
                              void* d_out, int out_size, void* d_ws, size_t ws_size,
                              hipStream_t stream) {
  (void)in_sizes; (void)n_in; (void)out_size; (void)ws_size;
  const float* x  = (const float*)d_in[0];
  const float* wq = (const float*)d_in[1];
  const float* bq = (const float*)d_in[2];
  const float* wk = (const float*)d_in[3];
  const float* bk = (const float*)d_in[4];
  const float* wv = (const float*)d_in[5];
  const float* bv = (const float*)d_in[6];
  const float* al = (const float*)d_in[7];
  float* out = (float*)d_out;

  // workspace: Qb 4MB | Kb 4MB | Vb 32MB (bf16)
  __bf16* Qb = (__bf16*)d_ws;
  __bf16* Kb = Qb + (size_t)kBATCH * kN * kCQK;
  __bf16* Vb = Kb + (size_t)kBATCH * kN * kCQK;

  hipError_t e;
  e = hipFuncSetAttribute((const void*)qkv_kernel,
                          hipFuncAttributeMaxDynamicSharedMemorySize, 32768);
  (void)e;
  e = hipFuncSetAttribute((const void*)attn_kernel,
                          hipFuncAttributeMaxDynamicSharedMemorySize, 156160);
  (void)e;

  qkv_kernel<<<dim3(kN / 32, kBATCH), 256, 32768, stream>>>(
      x, wq, bq, wk, bk, wv, bv, Qb, Kb, Vb);
  attn_kernel<<<dim3(kN / 64, kBATCH), 256, 156160, stream>>>(
      x, Qb, Kb, Vb, al, out);
}
